// ShiftedWindowAttentionC2V_39702677684407
// MI455X (gfx1250) — matrix operand order repaired and checked by reading, not compiled, not run on hardware
//
#include <hip/hip_runtime.h>
#include <cstdint>
#include <cstddef>

// ======================================================================
// Shifted-window attention block for MI455X (gfx1250, wave32, WMMA).
// Heavy GEMMs in bf16 via v_wmma_f32_16x16x32_bf16; double-buffered LDS
// tile staging via global_load_async_to_lds_b128 (ASYNCcnt) so the DMA
// overlaps WMMA issue; one s_barrier per K-step.
// ======================================================================

typedef __attribute__((ext_vector_type(16))) __bf16 v16bf;
typedef __attribute__((ext_vector_type(8)))  float  v8f;

#define DEV __device__ __forceinline__

// --------- problem constants ---------
#define BATCH   32
#define DC      512
#define DV      1024
#define NHEAD   16
#define HD      64
#define GRID_H  28
#define WS7     7
#define NTOK    49          // tokens per window
#define MTOT    25088       // BATCH * 784 = 512 windows * 49 tokens

// --------- CDNA5 async copy: 16B global -> LDS (ASYNCcnt) ---------
DEV void async_copy16(void* lds_dst, const void* gsrc) {
  unsigned loff = (unsigned)(uintptr_t)lds_dst;  // LDS aperture: addr[31:0] = LDS offset
  asm volatile("global_load_async_to_lds_b128 %0, %1, off"
               :: "v"(loff), "v"(gsrc) : "memory");
}
DEV void async_wait0() { asm volatile("s_wait_asynccnt 0" ::: "memory"); }

// --------- WMMA bf16 fragment loaders (wave32 ISA layouts) ---------
// A 16x32 (MxK): lane L holds row M=L&15; lanes 0-15 K in {0..7,16..23},
// lanes 16-31 K in {8..15,24..31}; 2 bf16 per VGPR.
DEV v16bf load_a_frag(const __bf16* base, int lane, int stride) {
  int m  = lane & 15;
  int kb = (lane < 16) ? 0 : 8;
  const unsigned* row = (const unsigned*)(base + m * stride);
  union { v16bf v; unsigned u[8]; } r;
#pragma unroll
  for (int q = 0; q < 8; ++q) {
    int k = (q < 4) ? (kb + 2 * q) : (16 + kb + 2 * (q - 4));
    r.u[q] = row[k >> 1];
  }
  return r.v;
}
// B 32x16 (KxN): lane L holds column N=L&15; lanes 0-15 K in {0..7,16..23},
// lanes 16-31 K in {8..15,24..31}. LDS holds B column-major: Bs[n*stride + k].
DEV v16bf load_b_frag(const __bf16* base, int lane, int stride) {
  int n  = lane & 15;
  int kb = (lane < 16) ? 0 : 8;
  const unsigned* col = (const unsigned*)(base + n * stride);
  union { v16bf v; unsigned u[8]; } r;
#pragma unroll
  for (int q = 0; q < 8; ++q) r.u[q] = col[((q < 4) ? (kb + 2 * q) : (16 + kb + 2 * (q - 4))) >> 1];
  return r.v;
}

DEV v8f wmma_bf16(v16bf a, v16bf b, v8f c) {
  return __builtin_amdgcn_wmma_f32_16x16x32_bf16(
      /*neg_a=*/false, a, /*neg_b=*/false, b,
      /*c_mod=*/(short)0, c, /*reuse_a=*/false, /*reuse_b=*/false);
}

// window-token index -> (batch, hw in 28x28 grid), undoing roll(-3,-3)
DEV int wtok_to_hw(int m, int& bout) {
  int win = m / NTOK, tok = m - win * NTOK;
  bout = win >> 4;
  int wi = win & 15;
  int wr = wi >> 2, wc = wi & 3;
  int tr = tok / WS7, tc = tok - tr * WS7;
  int r = wr * WS7 + tr, c = wc * WS7 + tc;
  int rr = r + 3; if (rr >= GRID_H) rr -= GRID_H;
  int cc = c + 3; if (cc >= GRID_H) cc -= GRID_H;
  return rr * GRID_H + cc;
}

// ======================================================================
// Weight repack kernels (f32 -> bf16, N-major so GEMM B columns are
// contiguous in K).
// ======================================================================
__global__ __launch_bounds__(256)
void repack_conv_w(const float* __restrict__ w, __bf16* __restrict__ wt) {
  int e = blockIdx.x * 256 + threadIdx.x;          // e = o*2048 + k
  if (e >= DV * 2048) return;
  int o = e >> 11, k = e & 2047;
  int qc = k >> 9, ci = k & 511;
  int kh = qc >> 1, kw = qc & 1;
  wt[e] = (__bf16)w[((o * DC + ci) * 2 + kh) * 2 + kw];
}

__global__ __launch_bounds__(256)
void repack_t(const float* __restrict__ w, __bf16* __restrict__ wt,
              int K, int N) {                      // w: KxN -> wt[n*K + k]
  int e = blockIdx.x * 256 + threadIdx.x;
  if (e >= K * N) return;
  int n = e / K, k = e - n * K;
  wt[e] = (__bf16)w[(size_t)k * N + n];
}

// ======================================================================
// LayerNorm kernels
// ======================================================================
__global__ __launch_bounds__(256)
void ln1_kernel(const float* __restrict__ x, const float* __restrict__ g,
                const float* __restrict__ b, __bf16* __restrict__ xln) {
  const int row = blockIdx.x;                      // B*3136 rows, 512 feats
  const int t = threadIdx.x;
  const float* xr = x + (size_t)row * DC;
  float v0 = xr[t], v1 = xr[t + 256];
  __shared__ float red[256];
  red[t] = v0 + v1; __syncthreads();
  for (int o = 128; o > 0; o >>= 1) { if (t < o) red[t] += red[t + o]; __syncthreads(); }
  float mu = red[0] * (1.0f / DC); __syncthreads();
  float d0 = v0 - mu, d1 = v1 - mu;
  red[t] = d0 * d0 + d1 * d1; __syncthreads();
  for (int o = 128; o > 0; o >>= 1) { if (t < o) red[t] += red[t + o]; __syncthreads(); }
  float rs = rsqrtf(red[0] * (1.0f / DC) + 1e-6f);
  __bf16* out = xln + (size_t)row * DC;
  out[t]       = (__bf16)(d0 * rs * g[t] + b[t]);
  out[t + 256] = (__bf16)(d1 * rs * g[t + 256] + b[t + 256]);
}

// LN2 + roll(-3,-3) + window partition gather, output in window-token order
__global__ __launch_bounds__(256)
void ln2_gather_kernel(const float* __restrict__ h, const float* __restrict__ g,
                       const float* __restrict__ b, __bf16* __restrict__ xw) {
  const int m = blockIdx.x;                        // MTOT rows, 1024 feats
  const int t = threadIdx.x;
  int bb; int hw = wtok_to_hw(m, bb);
  const float* hr = h + ((size_t)bb * 784 + hw) * DV;
  float v[4];
#pragma unroll
  for (int j = 0; j < 4; ++j) v[j] = hr[t + 256 * j];
  __shared__ float red[256];
  red[t] = v[0] + v[1] + v[2] + v[3]; __syncthreads();
  for (int o = 128; o > 0; o >>= 1) { if (t < o) red[t] += red[t + o]; __syncthreads(); }
  float mu = red[0] * (1.0f / DV); __syncthreads();
  float ss = 0.f;
#pragma unroll
  for (int j = 0; j < 4; ++j) { v[j] -= mu; ss += v[j] * v[j]; }
  red[t] = ss; __syncthreads();
  for (int o = 128; o > 0; o >>= 1) { if (t < o) red[t] += red[t + o]; __syncthreads(); }
  float rs = rsqrtf(red[0] * (1.0f / DV) + 1e-5f);
  __bf16* out = xw + (size_t)m * DV;
#pragma unroll
  for (int j = 0; j < 4; ++j) {
    int idx = t + 256 * j;
    out[idx] = (__bf16)(v[j] * rs * g[idx] + b[idx]);
  }
}

// ======================================================================
// Generic bf16 WMMA GEMM: C[M,N] = A[M,K] @ B[K,N] (+epilogue).
// Block tile 128x128, BK=32, 256 threads = 8 waves (4 along M x 2 along N),
// each wave: 2x4 = 8 tiles of 16x16.
// Double-buffered async global->LDS staging: stage tile k+1 while issuing
// WMMAs for tile k; one barrier per K-step.
// MODE 0: conv im2col gather A, write f32 shortcut.
// MODE 1: plain A, +bias, scale q by 0.125, write bf16 qkv.
// MODE 2: plain A, +bias, window-reverse scatter + residual -> f32 out.
// ======================================================================
template <int MODE, int KDIM, int NTOT2>
__global__ __launch_bounds__(256)
void gemm_bf16_wmma(const __bf16* __restrict__ A, const __bf16* __restrict__ BT,
                    const float* __restrict__ bias, __bf16* __restrict__ outb,
                    float* __restrict__ outf, const float* __restrict__ shortcut) {
  __shared__ __align__(16) __bf16 As[2][128 * 32];
  __shared__ __align__(16) __bf16 Bs[2][128 * 32];
  const int tid = threadIdx.x;
  const int n0 = blockIdx.x * 128;
  const int m0 = blockIdx.y * 128;
  const int lane = tid & 31, wave = tid >> 5;
  const int wm = wave & 3, wn = wave >> 2;

  // ---- hoisted per-thread staging coordinates (2 x 16B chunks each for A,B)
  int arow[2], apart[2], bcol[2], bpart[2];
  int abI[2], aoh[2], aow[2];                       // MODE 0 im2col coords
  const __bf16* asrc0[2];                           // plain-mode A chunk base
  const __bf16* bsrc0[2];                           // B chunk base
#pragma unroll
  for (int it = 0; it < 2; ++it) {
    int cc = tid + it * 256;                        // 512 chunks of 16B
    arow[it] = cc >> 2; apart[it] = cc & 3;
    bcol[it] = cc >> 2; bpart[it] = cc & 3;
    if (MODE == 0) {
      int m = m0 + arow[it];
      int bI = m / 784, p = m - bI * 784;
      abI[it] = bI; aoh[it] = p / 28; aow[it] = p - aoh[it] * 28;
      asrc0[it] = A;                                // unused base
    } else {
      asrc0[it] = A + (size_t)(m0 + arow[it]) * KDIM + apart[it] * 8;
    }
    bsrc0[it] = BT + (size_t)(n0 + bcol[it]) * KDIM + bpart[it] * 8;
  }

  auto stage = [&](int buf, int kkn) {
#pragma unroll
    for (int it = 0; it < 2; ++it) {
      const __bf16* s;
      if (MODE == 0) {
        int qc = kkn >> 9, ci0 = kkn & 511;         // which of the 2x2 taps
        int kh = qc >> 1, kw = qc & 1;
        int pix = (2 * aoh[it] + kh) * 56 + (2 * aow[it] + kw);
        s = A + ((size_t)abI[it] * 3136 + pix) * DC + ci0 + apart[it] * 8;
      } else {
        s = asrc0[it] + kkn;
      }
      async_copy16(&As[buf][arow[it] * 32 + apart[it] * 8], s);
    }
#pragma unroll
    for (int it = 0; it < 2; ++it)
      async_copy16(&Bs[buf][bcol[it] * 32 + bpart[it] * 8], bsrc0[it] + kkn);
    __builtin_prefetch(bsrc0[0] + kkn + 64, 0, 0);  // global_prefetch_b8
  };

  const v8f vzero = {0.f, 0.f, 0.f, 0.f, 0.f, 0.f, 0.f, 0.f};
  v8f acc[2][4];
#pragma unroll
  for (int i = 0; i < 2; ++i)
#pragma unroll
    for (int j = 0; j < 4; ++j) acc[i][j] = vzero;

  stage(0, 0);                                      // prologue: tile 0
  int cur = 0;
  for (int kk = 0; kk < KDIM; kk += 32) {
    async_wait0();          // this wave's staged copies for buf[cur] landed
    __syncthreads();        // all waves' copies landed; all reads of buf[cur^1] done
    if (kk + 32 < KDIM) stage(cur ^ 1, kk + 32);    // DMA next tile during compute

    v16bf aF[2], bF[4];
#pragma unroll
    for (int ti = 0; ti < 2; ++ti)
      aF[ti] = load_a_frag(&As[cur][(wm * 32 + ti * 16) * 32], lane, 32);
#pragma unroll
    for (int tj = 0; tj < 4; ++tj)
      bF[tj] = load_b_frag(&Bs[cur][(wn * 64 + tj * 16) * 32], lane, 32);
#pragma unroll
    for (int ti = 0; ti < 2; ++ti)
#pragma unroll
      for (int tj = 0; tj < 4; ++tj)
        acc[ti][tj] = wmma_bf16(aF[ti], bF[tj], acc[ti][tj]);
    cur ^= 1;
  }

  // ---- epilogue ----
  const int nl = lane & 15;
  const int rhi = (lane < 16) ? 0 : 8;
#pragma unroll
  for (int ti = 0; ti < 2; ++ti) {
#pragma unroll
    for (int tj = 0; tj < 4; ++tj) {
      int gn = n0 + wn * 64 + tj * 16 + nl;
      float bn = bias[gn];
#pragma unroll
      for (int r = 0; r < 8; ++r) {
        int gm = m0 + wm * 32 + ti * 16 + rhi + r;
        float v = acc[ti][tj][r] + bn;
        if (MODE == 0) {
          outf[(size_t)gm * DV + gn] = v;
        } else if (MODE == 1) {
          if (gn < DV) v *= 0.125f;                 // SCALE = 64^-0.5
          outb[(size_t)gm * (3 * DV) + gn] = (__bf16)v;
        } else {
          int bI; int hw = wtok_to_hw(gm, bI);
          size_t d = ((size_t)bI * 784 + hw) * DV + gn;
          outf[d] = shortcut[d] + v;
        }
      }
    }
  }
}

// ======================================================================
// Attention: one block per (window, head), 128 threads = 4 waves.
// S = Q@K^T via WMMA (64x64 padded), fused bias+mask+softmax, O = P@V.
// ======================================================================
__global__ __launch_bounds__(128)
void attn_wmma(const __bf16* __restrict__ qkv, const float* __restrict__ bias_table,
               __bf16* __restrict__ attn_out) {
  __shared__ __align__(16) __bf16 Qs[64 * 64];
  __shared__ __align__(16) __bf16 Ks[64 * 64];
  __shared__ __align__(16) __bf16 Vt[64 * 64];     // transposed: [feat][tok]
  __shared__ __align__(16) __bf16 Ps[64 * 64];
  __shared__ float Sf[64 * 64];
  const int win = blockIdx.x, head = blockIdx.y;
  const int tid = threadIdx.x, lane = tid & 31, wave = tid >> 5;
  const size_t base = (size_t)win * NTOK * (3 * DV) + head * HD;

  // load Q and K rows (zero-pad rows 49..63)
  for (int cc = tid; cc < 512; cc += 128) {
    int row = cc >> 3, part = cc & 7;
    uint4 zq = {0u, 0u, 0u, 0u}, zk = {0u, 0u, 0u, 0u};
    if (row < NTOK) {
      const __bf16* q = qkv + base + (size_t)row * (3 * DV) + part * 8;
      zq = *(const uint4*)q;
      zk = *(const uint4*)(q + DV);
    }
    *(uint4*)&Qs[row * 64 + part * 8] = zq;
    *(uint4*)&Ks[row * 64 + part * 8] = zk;
  }
  // load V transposed
  for (int e = tid; e < 4096; e += 128) {
    int f = e & 63, t = e >> 6;
    __bf16 v = (__bf16)0.0f;
    if (t < NTOK) v = qkv[base + (size_t)t * (3 * DV) + 2 * DV + f];
    Vt[f * 64 + t] = v;
  }
  __syncthreads();

  const v8f vzero = {0.f, 0.f, 0.f, 0.f, 0.f, 0.f, 0.f, 0.f};
  const int nl = lane & 15, rhi = (lane < 16) ? 0 : 8;

  // S = Q @ K^T  (K-dim = 64 features = 2 WMMA steps)
  v8f sacc[4];
#pragma unroll
  for (int j = 0; j < 4; ++j) sacc[j] = vzero;
#pragma unroll
  for (int s = 0; s < 2; ++s) {
    v16bf a = load_a_frag(&Qs[(wave * 16) * 64 + 32 * s], lane, 64);
#pragma unroll
    for (int j = 0; j < 4; ++j)
      sacc[j] = wmma_bf16(a, load_b_frag(&Ks[(j * 16) * 64 + 32 * s], lane, 64), sacc[j]);
  }
#pragma unroll
  for (int j = 0; j < 4; ++j)
#pragma unroll
    for (int r = 0; r < 8; ++r)
      Sf[(wave * 16 + rhi + r) * 64 + j * 16 + nl] = sacc[j][r];
  __syncthreads();

  // fused rel-pos bias + shift mask + softmax (one thread per row)
  if (tid < 64) {
    int i = tid;
    if (i < NTOK) {
      int wi = win & 15, wr = wi >> 2, wc = wi & 3;
      int ri = i / WS7, ci = i - ri * WS7;
      int r_ = wr * WS7 + ri, c_ = wc * WS7 + ci;
      int regi = ((r_ < 21) ? 0 : (r_ < 25 ? 1 : 2)) * 3 +
                 ((c_ < 21) ? 0 : (c_ < 25 ? 1 : 2));
      float srow[NTOK];
      float mx = -1e30f;
      for (int j = 0; j < NTOK; ++j) {
        int rj = j / WS7, cj = j - rj * WS7;
        int bi = (ri - rj + 6) * 13 + (ci - cj + 6);
        float s = Sf[i * 64 + j] + bias_table[bi * NHEAD + head];
        int r2 = wr * WS7 + rj, c2 = wc * WS7 + cj;
        int regj = ((r2 < 21) ? 0 : (r2 < 25 ? 1 : 2)) * 3 +
                   ((c2 < 21) ? 0 : (c2 < 25 ? 1 : 2));
        if (regj != regi) s -= 100.0f;
        srow[j] = s;
        mx = fmaxf(mx, s);
      }
      float sum = 0.f;
      for (int j = 0; j < NTOK; ++j) { float e = __expf(srow[j] - mx); srow[j] = e; sum += e; }
      float inv = 1.0f / sum;
      for (int j = 0; j < NTOK; ++j) Ps[i * 64 + j] = (__bf16)(srow[j] * inv);
      for (int j = NTOK; j < 64; ++j) Ps[i * 64 + j] = (__bf16)0.0f;
    } else {
      for (int j = 0; j < 64; ++j) Ps[i * 64 + j] = (__bf16)0.0f;
    }
  }
  __syncthreads();

  // O = P @ V  (K-dim = 64 tokens = 2 WMMA steps; Vt is [feat][tok])
  v8f oacc[4];
#pragma unroll
  for (int j = 0; j < 4; ++j) oacc[j] = vzero;
#pragma unroll
  for (int s = 0; s < 2; ++s) {
    v16bf a = load_a_frag(&Ps[(wave * 16) * 64 + 32 * s], lane, 64);
#pragma unroll
    for (int j = 0; j < 4; ++j)
      oacc[j] = wmma_bf16(a, load_b_frag(&Vt[(j * 16) * 64 + 32 * s], lane, 64), oacc[j]);
  }
#pragma unroll
  for (int j = 0; j < 4; ++j)
#pragma unroll
    for (int r = 0; r < 8; ++r) {
      int row = wave * 16 + rhi + r;
      if (row < NTOK)
        attn_out[((size_t)win * NTOK + row) * DV + head * HD + j * 16 + nl] =
            (__bf16)oacc[j][r];
    }
}

// ======================================================================
// Host launcher
// ======================================================================
extern "C" void kernel_launch(void* const* d_in, const int* in_sizes, int n_in,
                              void* d_out, int out_size, void* d_ws, size_t ws_size,
                              hipStream_t stream) {
  (void)in_sizes; (void)n_in; (void)out_size;
  const float* x          = (const float*)d_in[0];
  const float* ln1_g      = (const float*)d_in[1];
  const float* ln1_b      = (const float*)d_in[2];
  const float* conv_w     = (const float*)d_in[3];
  const float* conv_b     = (const float*)d_in[4];
  const float* ln2_g      = (const float*)d_in[5];
  const float* ln2_b      = (const float*)d_in[6];
  const float* qkv_w      = (const float*)d_in[7];
  const float* qkv_b      = (const float*)d_in[8];
  const float* proj_w     = (const float*)d_in[9];
  const float* proj_b     = (const float*)d_in[10];
  const float* bias_table = (const float*)d_in[11];
  float* out = (float*)d_out;

  // workspace layout (bytes, 256-aligned)
  char* ws = (char*)d_ws;
  const size_t OFF_XLN   = 0;                  // bf16  32*3136*512
  const size_t OFF_WCONV = 102760448;          // bf16  1024*2048 (N-major)
  const size_t OFF_SHORT = 106954752;          // f32   25088*1024
  const size_t OFF_XW    = 209715200;          // bf16  25088*1024
  const size_t OFF_WQKV  = 261095424;          // bf16  3072*1024 (N-major)
  const size_t OFF_QKV   = 267386880;          // bf16  25088*3072
  const size_t OFF_ATTN  = 421527552;          // bf16  25088*1024
  const size_t OFF_WPROJ = 472907776;          // bf16  1024*1024 (N-major)
  const size_t WS_NEED   = 475004928;
  if (ws_size < WS_NEED) return;

  __bf16* xln     = (__bf16*)(ws + OFF_XLN);
  __bf16* wconvT  = (__bf16*)(ws + OFF_WCONV);
  float*  shortcf = (float*) (ws + OFF_SHORT);
  __bf16* xw      = (__bf16*)(ws + OFF_XW);
  __bf16* wqkvT   = (__bf16*)(ws + OFF_WQKV);
  __bf16* qkvB    = (__bf16*)(ws + OFF_QKV);
  __bf16* attnO   = (__bf16*)(ws + OFF_ATTN);
  __bf16* wprojT  = (__bf16*)(ws + OFF_WPROJ);

  // 1) weight repacks
  repack_conv_w<<<(DV * 2048 + 255) / 256, 256, 0, stream>>>(conv_w, wconvT);
  repack_t<<<(DV * 3 * DV + 255) / 256, 256, 0, stream>>>(qkv_w, wqkvT, DV, 3 * DV);
  repack_t<<<(DV * DV + 255) / 256, 256, 0, stream>>>(proj_w, wprojT, DV, DV);

  // 2) LN1 -> bf16
  ln1_kernel<<<BATCH * 3136, 256, 0, stream>>>(x, ln1_g, ln1_b, xln);

  // 3) conv pooling as im2col GEMM: M=25088, K=2048, N=1024 -> shortcut f32
  gemm_bf16_wmma<0, 2048, 1024><<<dim3(8, 196), 256, 0, stream>>>(
      xln, wconvT, conv_b, nullptr, shortcf, nullptr);

  // 4) LN2 + roll + window gather -> bf16 window-token-ordered activations
  ln2_gather_kernel<<<MTOT, 256, 0, stream>>>(shortcf, ln2_g, ln2_b, xw);

  // 5) QKV GEMM: M=25088, K=1024, N=3072 (+bias, q*scale) -> bf16
  gemm_bf16_wmma<1, 1024, 3072><<<dim3(24, 196), 256, 0, stream>>>(
      xw, wqkvT, qkv_b, qkvB, nullptr, nullptr);

  // 6) windowed attention (512 windows x 16 heads)
  attn_wmma<<<dim3(512, NHEAD), 128, 0, stream>>>(qkvB, bias_table, attnO);

  // 7) proj GEMM + window-reverse/roll-back scatter + residual -> f32 out
  gemm_bf16_wmma<2, 1024, 1024><<<dim3(8, 196), 256, 0, stream>>>(
      attnO, wprojT, proj_b, nullptr, out, shortcf);
}
